// MalConvLowMem_19447611916330
// MI455X (gfx1250) — matrix operand order repaired and checked by reading, not compiled, not run on hardware
//
#include <hip/hip_runtime.h>
#include <hip/hip_bf16.h>

typedef __attribute__((ext_vector_type(16))) __bf16 v16bf;
typedef __attribute__((ext_vector_type(8)))  float  v8f;

#define KC     4096      // contraction length = E*K = 8*512
#define C_DIM  128
#define BM     128
#define BN     64
#define BK     64
#define LDA    36        // dwords per LDS row: 32 (64 bf16) + 4 pad, multiple of 4 (16B align)
#define LDB    36

__device__ __forceinline__ unsigned f2bf(float f) {          // fp32 -> bf16 bits, RNE
  unsigned u = __float_as_uint(f);
  u += 0x7FFFu + ((u >> 16) & 1u);
  return u >> 16;
}

// pack two fp32 -> bf16x2: one add per value (round-half-up) + one v_perm_b32
__device__ __forceinline__ unsigned pkbf(float x, float y) {
  unsigned ux = __float_as_uint(x) + 0x8000u;
  unsigned uy = __float_as_uint(y) + 0x8000u;
  return __builtin_amdgcn_perm(uy, ux, 0x07060302u);  // {uy[31:16], ux[31:16]}
}

__device__ __forceinline__ unsigned flipf(float f) {         // order-preserving float->uint
  unsigned u = __float_as_uint(f);
  return (u & 0x80000000u) ? ~u : (u | 0x80000000u);
}
__device__ __forceinline__ float unflipf(unsigned u) {
  unsigned v = (u & 0x80000000u) ? (u & 0x7fffffffu) : ~u;
  return __uint_as_float(v);
}

struct U8x32 { uint4 lo, hi; };
__device__ __forceinline__ v16bf frag_ld(const unsigned* p0, const unsigned* p1) {
  U8x32 t;
  t.lo = *(const uint4*)p0;   // ds_load_b128
  t.hi = *(const uint4*)p1;   // ds_load_b128
  return __builtin_bit_cast(v16bf, t);
}

// ---- repack weights: Wt[2c+s][k*8+e] = Ws[c][e][k] (bf16); bias interleaved ----
__global__ void malconv_prep_w(const float* __restrict__ W1, const float* __restrict__ b1,
                               const float* __restrict__ W2, const float* __restrict__ b2,
                               unsigned short* __restrict__ Wt, float* __restrict__ biasw) {
  unsigned o = blockIdx.x * 256u + threadIdx.x;   // 0 .. 1048575
  unsigned n = o >> 12, j = o & 4095u;
  unsigned c = n >> 1, s = n & 1u;
  unsigned k = j >> 3, e = j & 7u;
  const float* W = s ? W2 : W1;
  Wt[o] = (unsigned short)f2bf(W[(c * 8u + e) * 512u + k]);
  if (o < 256u) biasw[o] = (o & 1u) ? b2[o >> 1] : b1[o >> 1];
}

__global__ void malconv_init_acc(unsigned* __restrict__ accum) {
  int i = blockIdx.x * 256 + threadIdx.x;
  if (i < 1024) accum[i] = 0u;                    // < flip of any real value
}

// ---- fused GEMM + bias + gate + max ----
__global__ __launch_bounds__(256)
void malconv_gemm_gate(const float* __restrict__ z, const unsigned short* __restrict__ Wt,
                       const float* __restrict__ biasw, unsigned* __restrict__ accum) {
  __shared__ unsigned As[BM * LDA];
  __shared__ unsigned Bs[BN * LDB];
  __shared__ unsigned red[32];

  const int tid  = threadIdx.x;
  const int m0   = blockIdx.y * BM;               // 128 time-windows, all in one batch
  const int n0   = blockIdx.x * BN;               // 64 interleaved channel columns
  const int w    = tid >> 5;                      // wave 0..7 (wave32)
  const int lane = tid & 31;
  const int l15  = lane & 15;
  const int half = lane >> 4;
  const int wm   = w & 3;                         // 4 waves along M
  const int wn   = w >> 2;                        // 2 waves along N

  v8f acc[2][2] = {};
  const unsigned* WtU = (const unsigned*)Wt;

  // per-thread staging coordinates
  const int arow = tid >> 4,  ac4 = tid & 15;     // A: 16 float4 per row
  const int brow = tid >> 3,  bc  = tid & 7;      // B: 8 uint4 per row

  const float*    zA = z   + (size_t)(m0 + arow) * KC + ac4 * 4;
  const unsigned* wB = WtU + (size_t)(n0 + brow) * (KC / 2) + bc * 4;

  // tile load: 2x b128 (B) + 8x b128 (A) into registers
  auto load_tile = [&](int ko, float4 (&va)[8], uint4 (&vb)[2]) {
    #pragma unroll
    for (int i = 0; i < 2; ++i)
      vb[i] = *(const uint4*)(wB + (size_t)i * 32 * (KC / 2) + (ko >> 1));
    #pragma unroll
    for (int i = 0; i < 8; ++i)
      va[i] = *(const float4*)(zA + (size_t)i * 16 * KC + ko);
  };
  // stage tile: B raw, A converted fp32 -> bf16x2
  auto stage_tile = [&](const float4 (&va)[8], const uint4 (&vb)[2]) {
    #pragma unroll
    for (int i = 0; i < 2; ++i)
      *(uint4*)&Bs[(brow + i * 32) * LDB + bc * 4] = vb[i];
    #pragma unroll
    for (int i = 0; i < 8; ++i)
      *(uint2*)&As[(arow + i * 16) * LDA + ac4 * 2] =
          make_uint2(pkbf(va[i].x, va[i].y), pkbf(va[i].z, va[i].w));
  };
  // 2 K-steps x 4 WMMA on the staged tile
  auto wmma_tile = [&]() {
    #pragma unroll
    for (int ks = 0; ks < BK; ks += 32) {
      const int au = (ks >> 1) + half * 4;        // A: K = half*8 + {0..7} and +16
      const int bu = (ks >> 1) + half * 4;        // B: K = half*8 + {0..7} and +16
      v16bf a0 = frag_ld(&As[(wm * 32      + l15) * LDA + au], &As[(wm * 32      + l15) * LDA + au + 8]);
      v16bf a1 = frag_ld(&As[(wm * 32 + 16 + l15) * LDA + au], &As[(wm * 32 + 16 + l15) * LDA + au + 8]);
      v16bf b0 = frag_ld(&Bs[(wn * 32      + l15) * LDB + bu], &Bs[(wn * 32      + l15) * LDB + bu + 8]);
      v16bf b1 = frag_ld(&Bs[(wn * 32 + 16 + l15) * LDB + bu], &Bs[(wn * 32 + 16 + l15) * LDB + bu + 8]);
      acc[0][0] = __builtin_amdgcn_wmma_f32_16x16x32_bf16(false, a0, false, b0, (short)0, acc[0][0], false, false);
      acc[0][1] = __builtin_amdgcn_wmma_f32_16x16x32_bf16(false, a0, false, b1, (short)0, acc[0][1], false, false);
      acc[1][0] = __builtin_amdgcn_wmma_f32_16x16x32_bf16(false, a1, false, b0, (short)0, acc[1][0], false, false);
      acc[1][1] = __builtin_amdgcn_wmma_f32_16x16x32_bf16(false, a1, false, b1, (short)0, acc[1][1], false, false);
    }
  };

  // software pipeline: loads for tile i+1 issue BEFORE the WMMA block of tile i,
  // so the 8-WMMA block hides global load latency in-wave.
  float4 va0[8], va1[8];
  uint4  vb0[2], vb1[2];
  load_tile(0, va0, vb0);

  for (int ko = 0; ko < KC; ko += 2 * BK) {
    stage_tile(va0, vb0);
    __syncthreads();
    load_tile(ko + BK, va1, vb1);                 // always in range (KC/BK even)
    {                                             // prefetch tile i+2
      int nko = ko + 2 * BK;
      if (nko < KC)
        __builtin_prefetch(z + (size_t)(m0 + (tid >> 1)) * KC + nko + (tid & 1) * 32, 0, 1);
    }
    wmma_tile();
    __syncthreads();

    stage_tile(va1, vb1);
    __syncthreads();
    if (ko + 2 * BK < KC) load_tile(ko + 2 * BK, va0, vb0);
    wmma_tile();
    __syncthreads();
  }

  // ---- epilogue: bias + gate + max over rows ----
  if (tid < 32) red[tid] = 0u;
  __syncthreads();

  const int colL0 = wn * 32 + l15;                // local column for ni=0
  const float bias0 = biasw[n0 + colL0];
  const float bias1 = biasw[n0 + colL0 + 16];
  const bool evenc = ((l15 & 1) == 0);            // even column = c1 (W1), partner = c2 (W2)

  #pragma unroll
  for (int ni = 0; ni < 2; ++ni) {
    const float bn = ni ? bias1 : bias0;
    float gmax = -3.4e38f;
    #pragma unroll
    for (int mi = 0; mi < 2; ++mi) {
      #pragma unroll
      for (int r = 0; r < 8; ++r) {
        float v = acc[mi][ni][r] + bn;            // biased value for own column
        float p = __shfl_xor(v, 1);               // partner column (same rows)
        float g = v * (1.0f / (1.0f + __expf(-p)));
        gmax = fmaxf(gmax, g);
      }
    }
    if (evenc) atomicMax(&red[(colL0 + ni * 16) >> 1], flipf(gmax));
  }
  __syncthreads();
  if (tid < 32) {
    int b = m0 >> 11;                             // 2048 windows per batch
    atomicMax(&accum[b * C_DIM + (n0 >> 1) + tid], red[tid]);
  }
}

__global__ void malconv_finalize(const unsigned* __restrict__ accum, float* __restrict__ out) {
  int i = blockIdx.x * 256 + threadIdx.x;
  if (i < 1024) out[i] = unflipf(accum[i]);
}

extern "C" void kernel_launch(void* const* d_in, const int* in_sizes, int n_in,
                              void* d_out, int out_size, void* d_ws, size_t ws_size,
                              hipStream_t stream) {
  (void)in_sizes; (void)n_in; (void)out_size; (void)ws_size;
  const float* z  = (const float*)d_in[0];
  const float* W1 = (const float*)d_in[1];
  const float* b1 = (const float*)d_in[2];
  const float* W2 = (const float*)d_in[3];
  const float* b2 = (const float*)d_in[4];
  float* out = (float*)d_out;

  // workspace layout: Wt bf16 [256][4096] (2 MB) | bias f32[256] | accum u32[1024]
  unsigned short* Wt  = (unsigned short*)d_ws;
  float* biasw        = (float*)((char*)d_ws + (size_t)256 * 4096 * 2);
  unsigned* accum     = (unsigned*)((char*)d_ws + (size_t)256 * 4096 * 2 + 256 * 4);

  malconv_prep_w<<<4096, 256, 0, stream>>>(W1, b1, W2, b2, Wt, biasw);
  malconv_init_acc<<<4, 256, 0, stream>>>(accum);
  malconv_gemm_gate<<<dim3(4, 128), 256, 0, stream>>>(z, Wt, biasw, accum);
  malconv_finalize<<<4, 256, 0, stream>>>(accum, out);
}
